// SLEGNNEncoder_59674275610639
// MI455X (gfx1250) — matrix operand order repaired and checked by reading, not compiled, not run on hardware
//
#include <hip/hip_runtime.h>
#include <hip/hip_bf16.h>

// ---------------- problem constants ----------------
constexpr int NN  = 100000;   // nodes (exactly 6250 * 16)
constexpr int EE  = 1600000;  // edges
constexpr int CIN = 128;      // input feature dim
constexpr int HH  = 256;      // hidden dim
constexpr int LL  = 3;        // layers
constexpr float EPSBN = 1e-5f;

// ---------------- workspace layout (bytes) ----------------
constexpr size_t OFF_HW   = 0;                                   // N*H f32
constexpr size_t OFF_AGG  = OFF_HW  + (size_t)NN * HH * 4;       // N*H f32 (also aliases xbf pre-loop)
constexpr size_t OFF_DEG  = OFF_AGG + (size_t)NN * HH * 4;       // N f32
constexpr size_t OFF_DINV = OFF_DEG + (size_t)NN * 4;            // N f32
constexpr size_t OFF_SUMS = OFF_DINV + (size_t)NN * 4;           // 2*H f32 (sum, sumsq)
constexpr size_t OFF_W0T  = OFF_SUMS + (size_t)2 * HH * 4;       // H*CIN bf16 (W0^T)
constexpr size_t OFF_WT   = OFF_W0T + (size_t)HH * CIN * 2;      // L*H*H bf16 (W[l]^T)
constexpr size_t OFF_HBF  = OFF_WT  + (size_t)LL * HH * HH * 2;  // N*H bf16 (h mirror)

// ---------------- WMMA vector types ----------------
typedef __attribute__((ext_vector_type(16))) __bf16       v16bf;
typedef __attribute__((ext_vector_type(8)))  float        v8f;
typedef __attribute__((ext_vector_type(8)))  unsigned int v8u;

// f32 -> bf16 round-to-nearest-even (bit pattern as ushort)
__device__ __forceinline__ unsigned short f2bf(float f) {
    unsigned int u = __float_as_uint(f);
    u += 0x7FFFu + ((u >> 16) & 1u);
    return (unsigned short)(u >> 16);
}
__device__ __forceinline__ unsigned int pk2(float lo, float hi) {
    return (unsigned int)f2bf(lo) | ((unsigned int)f2bf(hi) << 16);
}

// ---------------- degree kernels ----------------
__global__ void deg_init_kernel(float* __restrict__ deg) {
    int i = blockIdx.x * 256 + threadIdx.x;
    if (i < NN) deg[i] = 0.0f;
}

__global__ void deg_accum_kernel(const int* __restrict__ ei, float* __restrict__ deg) {
    int e = blockIdx.x * 256 + threadIdx.x;
    if (e < EE) {
        int d = ei[EE + e];                 // dst row of edge_index
        unsafeAtomicAdd(&deg[d], 1.0f);     // global_atomic_add_f32
    }
}

// ---------------- generic f32 -> bf16 convert (n divisible by 4) ----------------
__global__ void cvt_bf16_kernel(const float* __restrict__ src, unsigned short* __restrict__ dst,
                                int n4) {
    int i = blockIdx.x * 256 + threadIdx.x;
    if (i >= n4) return;
    float4 v = *(const float4*)(src + (size_t)i * 4);
    uint2 p = {pk2(v.x, v.y), pk2(v.z, v.w)};
    *(uint2*)(dst + (size_t)i * 4) = p;
}

// ---------------- weight transpose + bf16 convert ----------------
// W0t[n][k] = bf16(W0[k][n]);  Wt[l][n][k] = bf16(W[l][k][n])
__global__ void wconv_kernel(const float* __restrict__ W0, const float* __restrict__ W,
                             unsigned short* __restrict__ W0t, unsigned short* __restrict__ Wt) {
    const int total = HH * CIN + LL * HH * HH;
    for (int idx = blockIdx.x * 256 + threadIdx.x; idx < total; idx += gridDim.x * 256) {
        if (idx < HH * CIN) {
            int n = idx / CIN, k = idx % CIN;
            W0t[idx] = f2bf(W0[(size_t)k * HH + n]);
        } else {
            int o = idx - HH * CIN;
            int l = o / (HH * HH);
            int rem = o % (HH * HH);
            int n = rem / HH, k = rem % HH;
            Wt[o] = f2bf(W[(size_t)l * HH * HH + (size_t)k * HH + n]);
        }
    }
}

// ---------------- WMMA GEMM: C[M=NN, HH] = A[M, K] x Bt^T (+bias) ----------------
// One wave computes one 16x16 output tile; 8 waves/block cover 128 output columns.
// Abf: bf16 [M][K] row-major.  Bt: bf16 [HH][K] (i.e. B^T, K-contiguous per column).
// If Cbf != nullptr, also writes a bf16 mirror of C (used as next layer's A).
template <int K, bool HAS_BIAS>
__global__ __launch_bounds__(256)
void wmma_gemm_kernel(const unsigned short* __restrict__ Abf,
                      const unsigned short* __restrict__ Bt,
                      const float* __restrict__ bias,
                      float* __restrict__ C,
                      unsigned short* __restrict__ Cbf) {
    const int lane   = threadIdx.x & 31;
    const int wave   = threadIdx.x >> 5;
    const int tileM  = blockIdx.x;                 // 0..6249
    const int tileN  = blockIdx.y * 8 + wave;      // 0..15
    const int l15    = lane & 15;
    const int hiHalf = lane >> 4;                  // 0: lanes 0-15, 1: lanes 16-31

    const int row = tileM * 16 + l15;              // A row for this lane (M)
    const int col = tileN * 16 + l15;              // B column for this lane (N)

    const unsigned short* arow = Abf + (size_t)row * K;
    const unsigned short* brow = Bt  + (size_t)col * K;

    v8f acc = {};
    #pragma unroll
    for (int kk = 0; kk < K; kk += 32) {
        // ---- A operand: ISA 16-bit A 16x32 layout ----
        // lanes 0-15 : elems 0..7 = K kk+0..7,  elems 8..15 = K kk+16..23
        // lanes 16-31: elems 0..7 = K kk+8..15, elems 8..15 = K kk+24..31
        const int offA = kk + hiHalf * 8;
        uint4 a0 = *(const uint4*)(arow + offA);        // 8 bf16: K offA..offA+7
        uint4 a1 = *(const uint4*)(arow + offA + 16);   // 8 bf16: K offA+16..offA+23
        v8u pa = {a0.x, a0.y, a0.z, a0.w, a1.x, a1.y, a1.z, a1.w};
        v16bf av = __builtin_bit_cast(v16bf, pa);

        // ---- B operand: 32x16, same K order as A (lanes 0-15: K kk+0..7,16..23; lanes 16-31: K kk+8..15,24..31) ----
        const int offB = kk + hiHalf * 8;
        uint4 b0 = *(const uint4*)(brow + offB);
        uint4 b1 = *(const uint4*)(brow + offB + 16);
        v8u pb = {b0.x, b0.y, b0.z, b0.w, b1.x, b1.y, b1.z, b1.w};
        v16bf bv = __builtin_bit_cast(v16bf, pb);

        acc = __builtin_amdgcn_wmma_f32_16x16x32_bf16(
            /*neg_a=*/false, av, /*neg_b=*/false, bv,
            /*c_mod=*/(short)0, acc, /*reuse_a=*/false, /*reuse_b=*/false);
    }

    float bv_add = 0.0f;
    if (HAS_BIAS) bv_add = bias[col];

    // ---- C layout: VGPR r -> row (tileM*16 + hiHalf*8 + r), col = lane&15 ----
    const int rbase = tileM * 16 + hiHalf * 8;
    float* cp = C + (size_t)rbase * HH + col;
    #pragma unroll
    for (int r = 0; r < 8; ++r) {
        float v = acc[r] + bv_add;
        cp[(size_t)r * HH] = v;
        if (Cbf) Cbf[(size_t)(rbase + r) * HH + col] = f2bf(v);
    }
}

// ---------------- agg init: agg = (n_loops/deg)*hw + b[l]; also dinv + zero BN sums ----------------
__global__ void init_agg_kernel(const float* __restrict__ hw, const float* __restrict__ deg,
                                const float* __restrict__ bvec, float nloops,
                                float* __restrict__ agg, float* __restrict__ dinv,
                                float* __restrict__ sums) {
    if (blockIdx.x == 0) {                  // zero BN accumulators (sum, sumsq)
        sums[threadIdx.x] = 0.0f;
        sums[HH + threadIdx.x] = 0.0f;
    }
    int idx = blockIdx.x * 256 + threadIdx.x;   // one float4 per thread
    if (idx >= NN * (HH / 4)) return;
    int n  = idx >> 6;            // / (HH/4)
    int c4 = (idx & 63) * 4;
    float d  = deg[n] + nloops;
    float sc = nloops / d;
    if ((idx & 63) == 0) dinv[n] = rsqrtf(d);
    float4 hv = *(const float4*)(hw + (size_t)n * HH + c4);
    float4 bb = *(const float4*)(bvec + c4);
    float4 out = {sc * hv.x + bb.x, sc * hv.y + bb.y, sc * hv.z + bb.z, sc * hv.w + bb.w};
    *(float4*)(agg + (size_t)n * HH + c4) = out;
}

// ---------------- edge scatter: agg[dst] += dinv[src]*dinv[dst] * hw[src] ----------------
// 4 edges per 256-thread block; 64 threads/edge, float4 per thread.
__global__ void scatter_kernel(const int* __restrict__ ei, const float* __restrict__ hw,
                               const float* __restrict__ dinv, float* __restrict__ agg) {
    int e = blockIdx.x * 4 + (threadIdx.x >> 6);
    if (e >= EE) return;
    int s = ei[e];
    int d = ei[EE + e];
    float coef = dinv[s] * dinv[d];
    int c4 = (threadIdx.x & 63) * 4;
    float4 hv = *(const float4*)(hw + (size_t)s * HH + c4);
    float* ap = agg + (size_t)d * HH + c4;
    unsafeAtomicAdd(ap + 0, coef * hv.x);
    unsafeAtomicAdd(ap + 1, coef * hv.y);
    unsafeAtomicAdd(ap + 2, coef * hv.z);
    unsafeAtomicAdd(ap + 3, coef * hv.w);
}

// ---------------- BN reduce: per-channel sum & sumsq ----------------
__global__ void bn_reduce_kernel(const float* __restrict__ agg, float* __restrict__ sums) {
    int c = threadIdx.x;
    float s = 0.0f, s2 = 0.0f;
    for (int n = blockIdx.x; n < NN; n += gridDim.x) {
        float v = agg[(size_t)n * HH + c];
        s += v; s2 += v * v;
    }
    unsafeAtomicAdd(&sums[c], s);
    unsafeAtomicAdd(&sums[HH + c], s2);
}

// ---------------- BN apply + ReLU + residual; updates h in place + bf16 mirror ----------------
__global__ void bn_apply_kernel(const float* __restrict__ agg, const float* __restrict__ sums,
                                const float* __restrict__ gamma, const float* __restrict__ beta,
                                float* __restrict__ h, unsigned short* __restrict__ hbf) {
    int c = threadIdx.x;
    size_t n = blockIdx.x;
    const float invN = 1.0f / (float)NN;
    float mu  = sums[c] * invN;
    float var = sums[HH + c] * invN - mu * mu;
    float inv = rsqrtf(fmaxf(var, 0.0f) + EPSBN);
    float a = agg[n * HH + c];
    float y = gamma[c] * (a - mu) * inv + beta[c];
    float hn = fmaxf(y, 0.0f) + h[n * HH + c];
    h[n * HH + c] = hn;
    hbf[n * HH + c] = f2bf(hn);
}

// ---------------- launch ----------------
extern "C" void kernel_launch(void* const* d_in, const int* in_sizes, int n_in,
                              void* d_out, int out_size, void* d_ws, size_t ws_size,
                              hipStream_t stream) {
    const float* x     = (const float*)d_in[0];
    const int*   ei    = (const int*)  d_in[1];   // [2, E] (int32 under default JAX config)
    const float* W0    = (const float*)d_in[2];
    const float* b0    = (const float*)d_in[3];
    const float* W     = (const float*)d_in[4];
    const float* b     = (const float*)d_in[5];
    const float* gamma = (const float*)d_in[6];
    const float* beta  = (const float*)d_in[7];

    float* h = (float*)d_out;                     // running hidden state == final output
    char*  ws = (char*)d_ws;
    float* hw   = (float*)(ws + OFF_HW);
    float* agg  = (float*)(ws + OFF_AGG);
    float* deg  = (float*)(ws + OFF_DEG);
    float* dinv = (float*)(ws + OFF_DINV);
    float* sums = (float*)(ws + OFF_SUMS);
    unsigned short* W0t = (unsigned short*)(ws + OFF_W0T);
    unsigned short* Wt  = (unsigned short*)(ws + OFF_WT);
    unsigned short* hbf = (unsigned short*)(ws + OFF_HBF);
    unsigned short* xbf = (unsigned short*)(ws + OFF_AGG);  // alias: agg unused until layer loop

    // degree + weight/input prep
    deg_init_kernel<<<(NN + 255) / 256, 256, 0, stream>>>(deg);
    deg_accum_kernel<<<(EE + 255) / 256, 256, 0, stream>>>(ei, deg);
    wconv_kernel<<<512, 256, 0, stream>>>(W0, W, W0t, Wt);
    cvt_bf16_kernel<<<(NN * (CIN / 4) + 255) / 256, 256, 0, stream>>>(x, xbf, NN * (CIN / 4));

    // h = x @ W0 + b0  (also emits hbf for the next GEMM's A operand)
    wmma_gemm_kernel<CIN, true><<<dim3(6250, 2), 256, 0, stream>>>(xbf, W0t, b0, h, hbf);

    for (int l = 0; l < LL; ++l) {
        const float nloops = (float)(l + 1);
        // hw = h @ W[l]   (A from bf16 mirror; no bf16 C mirror needed)
        wmma_gemm_kernel<HH, false><<<dim3(6250, 2), 256, 0, stream>>>(
            hbf, Wt + (size_t)l * HH * HH, nullptr, hw, nullptr);
        // agg = (n_loops/deg)*hw + b[l]; dinv = rsqrt(deg + n_loops); zero BN sums
        init_agg_kernel<<<(NN * (HH / 4) + 255) / 256, 256, 0, stream>>>(
            hw, deg, b + (size_t)l * HH, nloops, agg, dinv, sums);
        // agg[dst] += dinv[src]*dinv[dst]*hw[src]
        scatter_kernel<<<(EE + 3) / 4, 256, 0, stream>>>(ei, hw, dinv, agg);
        // batch-norm statistics
        bn_reduce_kernel<<<1024, 256, 0, stream>>>(agg, sums);
        // h = relu(BN(agg)) + h  (also refresh bf16 mirror for next layer)
        bn_apply_kernel<<<NN, 256, 0, stream>>>(
            agg, sums, gamma + (size_t)l * HH, beta + (size_t)l * HH, h, hbf);
    }
}
